// BooleanReservoir_76175539962221
// MI455X (gfx1250) — matrix operand order repaired and checked by reading, not compiled, not run on hardware
//
#include <hip/hip_runtime.h>
#include <stdint.h>

#define N_NODES 50000
#define N_PAD   50176                 // 49 * 1024, >= N_NODES+1 (zero node lives in pad)
#define ZERO_NODE N_NODES
#define KADJ    8
#define MROWS   256
#define SSTEPS  200
#define NIN     16
#define NSLICE  1024
#define NSLICES (N_PAD / NSLICE)      // 49
#define NPB     8                     // nodes per block in step kernel

typedef __attribute__((ext_vector_type(16))) _Float16 v16h;
typedef __attribute__((ext_vector_type(8)))  float    v8f;

// ---------------- one-time packing kernels ----------------

__global__ void init_input_idx(int* __restrict__ inputIdx) {
    int i = blockIdx.x * blockDim.x + threadIdx.x;
    if (i < N_NODES) inputIdx[i] = -1;
}

__global__ void scatter_input_idx(const int* __restrict__ input_nodes,
                                  int* __restrict__ inputIdx) {
    int i = threadIdx.x;
    if (i < NIN) inputIdx[input_nodes[i]] = i;
}

// lutP[n*8+w] bit b = lut[n*256 + w*32 + b]
__global__ void pack_lut(const unsigned char* __restrict__ lut,
                         uint32_t* __restrict__ lutP) {
    int t = blockIdx.x * blockDim.x + threadIdx.x;
    if (t >= N_NODES * 8) return;
    const uint32_t* p = (const uint32_t*)lut + (size_t)t * 8;
    uint32_t word = 0;
#pragma unroll
    for (int j = 0; j < 8; ++j) {
        uint32_t v = p[j];
        word |= (v & 1u)          << (4 * j + 0);
        word |= ((v >> 8)  & 1u)  << (4 * j + 1);
        word |= ((v >> 16) & 1u)  << (4 * j + 2);
        word |= ((v >> 24) & 1u)  << (4 * j + 3);
    }
    lutP[t] = word;
}

// stateA[n*8+w] bit m' = init_state[(w*32+m')*N + n]
__global__ void pack_state(const unsigned char* __restrict__ init_state,
                           uint32_t* __restrict__ stateA) {
    int t = blockIdx.x * blockDim.x + threadIdx.x;
    if (t >= N_NODES * 8) return;
    int n = t >> 3, w = t & 7;
    uint32_t word = 0;
    for (int mp = 0; mp < 32; ++mp)
        word |= (uint32_t)(init_state[(size_t)(w * 32 + mp) * N_NODES + n] & 1) << mp;
    stateA[t] = word;
}

// zero the padded node slots of both state buffers (zero-node + WMMA padding)
__global__ void zero_state_pad(uint32_t* __restrict__ stateA,
                               uint32_t* __restrict__ stateB) {
    int t = blockIdx.x * blockDim.x + threadIdx.x;
    if (t >= (N_PAD - N_NODES) * 8) return;
    stateA[N_NODES * 8 + t] = 0u;
    stateB[N_NODES * 8 + t] = 0u;
}

// xpk[s*128 + i*8 + w] bit m' = x[((w*32+m')*S + s)*16 + i]
__global__ void pack_x(const unsigned char* __restrict__ x,
                       uint32_t* __restrict__ xpk) {
    int t = blockIdx.x * blockDim.x + threadIdx.x;
    if (t >= SSTEPS * 16 * 8) return;
    int s = t >> 7; int r = t & 127; int i = r >> 3; int w = r & 7;
    uint32_t word = 0;
    for (int mp = 0; mp < 32; ++mp) {
        int m = w * 32 + mp;
        word |= (uint32_t)(x[((size_t)m * SSTEPS + s) * 16 + i] & 1) << mp;
    }
    xpk[(size_t)s * 128 + i * 8 + w] = word;
}

// adjF: masked-off edges point at the always-zero padded node.
// kidx: input-node index of the target (-1 if none / masked).
__global__ void pack_edges(const int* __restrict__ adj,
                           const unsigned char* __restrict__ amask,
                           const int* __restrict__ inputIdx,
                           int* __restrict__ adjF, int* __restrict__ kidx) {
    int t = blockIdx.x * blockDim.x + threadIdx.x;
    if (t >= N_NODES * 8) return;
    int a = adj[t];
    int m = amask[t] & 1;
    adjF[t] = m ? a : ZERO_NODE;
    kidx[t] = m ? inputIdx[a] : -1;
}

// Bmat[c*N_PAD+n]: c0/c1 = f16 hi(W[d]), c2/c3 = f16 lo(W[d]), rest zero.
__global__ void pack_bmat(const float* __restrict__ W, _Float16* __restrict__ Bmat) {
    int t = blockIdx.x * blockDim.x + threadIdx.x;
    if (t >= 16 * N_PAD) return;
    int c = t / N_PAD, n = t - c * N_PAD;
    _Float16 v = (_Float16)0.0f;
    if (n < N_NODES) {
        if (c < 2) {
            v = (_Float16)W[c * N_NODES + n];
        } else if (c < 4) {
            float wv = W[(c - 2) * N_NODES + n];
            _Float16 h = (_Float16)wv;
            v = (_Float16)(wv - (float)h);
        }
    }
    Bmat[t] = v;
}

// ---------------- main reservoir step ----------------
// Block = 8 waves; wave w handles 32-row word w for NPB nodes. All gather
// addresses are wave-uniform -> SMEM s_loads; only bit math is vector.
__global__ __launch_bounds__(256) void step_kernel(
    const uint32_t* __restrict__ cur, uint32_t* __restrict__ nxt,
    const int* __restrict__ adjF, const int* __restrict__ kidx,
    const uint32_t* __restrict__ lutP,
    const uint32_t* __restrict__ xpk /* [16][8] for this step */) {
    int w = __builtin_amdgcn_readfirstlane((int)(threadIdx.x >> 5));
    int l = threadIdx.x & 31;
    int n0 = blockIdx.x * NPB;
#pragma unroll 2
    for (int j = 0; j < NPB; ++j) {
        int n = n0 + j;
        uint32_t idx = 0;
#pragma unroll
        for (int k = 0; k < KADJ; ++k) {
            int a = adjF[n * 8 + k];            // s_load
            uint32_t wk = cur[a * 8 + w];       // s_load (uniform addr)
            int ii = kidx[n * 8 + k];           // s_load
            if (ii >= 0) wk ^= xpk[ii * 8 + w]; // scalar branch, rare
            idx |= ((wk >> l) & 1u) << k;       // per-lane bit
        }
        const uint32_t* lp = lutP + (size_t)n * 8;
        uint32_t l0 = lp[0], l1 = lp[1], l2 = lp[2], l3 = lp[3];
        uint32_t l4 = lp[4], l5 = lp[5], l6 = lp[6], l7 = lp[7];
        uint32_t w01 = (idx & 0x20u) ? l1 : l0;
        uint32_t w23 = (idx & 0x20u) ? l3 : l2;
        uint32_t w45 = (idx & 0x20u) ? l5 : l4;
        uint32_t w67 = (idx & 0x20u) ? l7 : l6;
        uint32_t w03 = (idx & 0x40u) ? w23 : w01;
        uint32_t w47 = (idx & 0x40u) ? w67 : w45;
        uint32_t wf  = (idx & 0x80u) ? w47 : w03;
        uint32_t bit = (wf >> (idx & 31u)) & 1u;
        uint32_t word = __builtin_amdgcn_ballot_w32(bit != 0);
        if (l == 0) nxt[n * 8 + w] = word;
    }
}

// ---------------- WMMA projection ----------------
__global__ __launch_bounds__(32) void proj_partial(
    const uint32_t* __restrict__ state, const _Float16* __restrict__ Bmat,
    float* __restrict__ partials) {
    int tile  = blockIdx.x;          // 0..15
    int slice = blockIdx.y;          // 0..NSLICES-1
    int l = threadIdx.x;
    int col = l & 15;
    int hiHalf = l >> 4;             // 0 or 1
    int m = tile * 16 + col;         // A row for this lane
    int wordA = m >> 5, bitA = m & 31;
    const _Float16* bp = Bmat + (size_t)col * N_PAD + hiHalf * 16;
    v8f acc = {};
    int n0 = slice * NSLICE;
    for (int nb = n0; nb < n0 + NSLICE; nb += 32) {
        v16h a;
#pragma unroll
        for (int e = 0; e < 16; ++e) {
            // A layout: K=e (+16 for hi lanes)
            int ka = e + hiHalf * 16;
            uint32_t bit = (state[(size_t)(nb + ka) * 8 + wordA] >> bitA) & 1u;
            a[e] = (_Float16)(float)bit;
        }
        v16h b = *(const v16h*)(bp + nb);   // B layout: K=e (+16 for hi lanes)
        acc = __builtin_amdgcn_wmma_f32_16x16x32_f16(
            false, a, false, b, (short)0, acc, false, false);
    }
    if (col < 4) {
#pragma unroll
        for (int j = 0; j < 8; ++j) {
            int row = tile * 16 + j + hiHalf * 8;
            partials[((size_t)slice * MROWS + row) * 4 + col] = acc[j];
        }
    }
}

__global__ void proj_reduce(const float* __restrict__ partials,
                            const float* __restrict__ bias,
                            float* __restrict__ out) {
    int t = blockIdx.x * blockDim.x + threadIdx.x;
    if (t >= MROWS * 2) return;
    int m = t >> 1, d = t & 1;
    float s = bias[d];
    for (int sl = 0; sl < NSLICES; ++sl) {       // fixed order -> deterministic
        s += partials[((size_t)sl * MROWS + m) * 4 + d];
        s += partials[((size_t)sl * MROWS + m) * 4 + d + 2];
    }
    out[t] = s;
}

// ---------------- host launch ----------------
extern "C" void kernel_launch(void* const* d_in, const int* in_sizes, int n_in,
                              void* d_out, int out_size, void* d_ws, size_t ws_size,
                              hipStream_t stream) {
    const unsigned char* x      = (const unsigned char*)d_in[0];
    const int*           adj    = (const int*)d_in[1];
    const unsigned char* amask  = (const unsigned char*)d_in[2];
    const int*           inodes = (const int*)d_in[3];
    const unsigned char* lut    = (const unsigned char*)d_in[4];
    const unsigned char* istate = (const unsigned char*)d_in[5];
    const float*         W      = (const float*)d_in[6];
    const float*         bias   = (const float*)d_in[7];
    float* out = (float*)d_out;

    char* ws = (char*)d_ws;
    size_t off = 0;
    #define WS_ALLOC(ptrty, name, bytes) \
        ptrty name = (ptrty)(ws + off); off = (off + (size_t)(bytes) + 255) & ~(size_t)255;
    WS_ALLOC(uint32_t*, stateA,  (size_t)N_PAD * 8 * 4)
    WS_ALLOC(uint32_t*, stateB,  (size_t)N_PAD * 8 * 4)
    WS_ALLOC(uint32_t*, lutP,    (size_t)N_NODES * 8 * 4)
    WS_ALLOC(uint32_t*, xpk,     (size_t)SSTEPS * 16 * 8 * 4)
    WS_ALLOC(int*,      inputIdx,(size_t)N_NODES * 4)
    WS_ALLOC(int*,      adjF,    (size_t)N_NODES * 8 * 4)
    WS_ALLOC(int*,      kidx,    (size_t)N_NODES * 8 * 4)
    WS_ALLOC(_Float16*, Bmat,    (size_t)16 * N_PAD * 2)
    WS_ALLOC(float*,    partials,(size_t)NSLICES * MROWS * 4 * 4)
    #undef WS_ALLOC
    (void)ws_size; (void)in_sizes; (void)n_in; (void)out_size;

    const int TPB = 256;
    init_input_idx<<<(N_NODES + TPB - 1) / TPB, TPB, 0, stream>>>(inputIdx);
    scatter_input_idx<<<1, 32, 0, stream>>>(inodes, inputIdx);
    pack_lut<<<(N_NODES * 8 + TPB - 1) / TPB, TPB, 0, stream>>>(lut, lutP);
    pack_state<<<(N_NODES * 8 + TPB - 1) / TPB, TPB, 0, stream>>>(istate, stateA);
    zero_state_pad<<<((N_PAD - N_NODES) * 8 + TPB - 1) / TPB, TPB, 0, stream>>>(stateA, stateB);
    pack_x<<<(SSTEPS * 16 * 8 + TPB - 1) / TPB, TPB, 0, stream>>>(x, xpk);
    pack_edges<<<(N_NODES * 8 + TPB - 1) / TPB, TPB, 0, stream>>>(adj, amask, inputIdx, adjF, kidx);
    pack_bmat<<<(16 * N_PAD + TPB - 1) / TPB, TPB, 0, stream>>>(W, Bmat);

    uint32_t* cur = stateA;
    uint32_t* nxt = stateB;
    for (int s = 0; s < SSTEPS; ++s) {
        step_kernel<<<N_NODES / NPB, 256, 0, stream>>>(
            cur, nxt, adjF, kidx, lutP, xpk + (size_t)s * 128);
        uint32_t* t = cur; cur = nxt; nxt = t;
    }

    proj_partial<<<dim3(16, NSLICES), 32, 0, stream>>>(cur, Bmat, partials);
    proj_reduce<<<1, 512, 0, stream>>>(partials, bias, out);
}
